// MoEGate_46420006535175
// MI455X (gfx1250) — matrix operand order repaired and checked by reading, not compiled, not run on hardware
//
#include <hip/hip_runtime.h>
#include <hip/hip_bf16.h>

// DeepSeek-V3 MoE gate, fused, bf16-split GEMM (3x V_WMMA_F32_16X16X32_BF16
// ~= exact f32) + sigmoid + bias + grouped top-k.
// 32 M-rows per block (2 M-tiles per wave) to halve weight L2 traffic and
// give the WMMA pipe 4 independent accumulation chains.
// d_ws: wh[256*7168] bf16 bits, wl[256*7168] bf16 bits (split weights)
// out: [T*8] indices (as float) followed by [T*8] weights.

typedef __attribute__((ext_vector_type(8)))  __bf16 v8bf;
typedef __attribute__((ext_vector_type(16))) __bf16 v16bf;
typedef __attribute__((ext_vector_type(8)))  float  v8f;

#define HDIM 7168
#define NEXP 256
#define KC 128
#define MROWS 32
#define A_STRIDE 136    // bf16 elems per LDS row: 272B, conflict-free b128 reads
#define SC_STRIDE 260   // conflict-free cross-row score reads

static __device__ __forceinline__ unsigned short f32_to_bf16_rne(float f) {
  unsigned u = __float_as_uint(f);
  unsigned r = u + 0x7FFFu + ((u >> 16) & 1u);
  return (unsigned short)(r >> 16);
}
static __device__ __forceinline__ float bf16_bits_to_f32(unsigned short h) {
  return __uint_as_float(((unsigned)h) << 16);
}
static __device__ __forceinline__ v16bf cat8(v8bf lo, v8bf hi) {
  return __builtin_shufflevector(lo, hi, 0, 1, 2, 3, 4, 5, 6, 7,
                                 8, 9, 10, 11, 12, 13, 14, 15);
}
static __device__ __forceinline__ v8bf ldg8(const unsigned short* base,
                                            unsigned off) {
  return *reinterpret_cast<const v8bf*>(base + off);
}
static __device__ __forceinline__ v8bf lds8(const unsigned short* p) {
  return *reinterpret_cast<const v8bf*>(p);
}
static __device__ __forceinline__ v8f wmma_bf16(v16bf a, v16bf b, v8f c) {
  return __builtin_amdgcn_wmma_f32_16x16x32_bf16(false, a, false, b,
                                                 (short)0, c, false, false);
}

// ---- prep: split f32 weight into bf16 hi/lo pair (RNE) ----
__global__ __launch_bounds__(256) void split_w_kernel(
    const float* __restrict__ w, unsigned short* __restrict__ wh,
    unsigned short* __restrict__ wl) {
  size_t i = ((size_t)blockIdx.x * 256 + threadIdx.x) * 4;
  float4 v = *reinterpret_cast<const float4*>(w + i);
  float f[4] = {v.x, v.y, v.z, v.w};
  unsigned short h[4], l[4];
#pragma unroll
  for (int j = 0; j < 4; ++j) {
    h[j] = f32_to_bf16_rne(f[j]);
    l[j] = f32_to_bf16_rne(f[j] - bf16_bits_to_f32(h[j]));
  }
  uint2 H2 = {(unsigned)h[0] | ((unsigned)h[1] << 16),
              (unsigned)h[2] | ((unsigned)h[3] << 16)};
  uint2 L2 = {(unsigned)l[0] | ((unsigned)l[1] << 16),
              (unsigned)l[2] | ((unsigned)l[3] << 16)};
  *reinterpret_cast<uint2*>(wh + i) = H2;
  *reinterpret_cast<uint2*>(wl + i) = L2;
}

// ---- main fused kernel ----
__global__ __launch_bounds__(256) void moe_gate_kernel(
    const float* __restrict__ x, const unsigned short* __restrict__ wh,
    const unsigned short* __restrict__ wl, const float* __restrict__ bias,
    float* __restrict__ outIdx, float* __restrict__ outW)
{
  // A-staging (double buffered bf16 hi/lo) and the score tile alias the same
  // LDS region; phases are separated by barriers.
  __shared__ __align__(16) unsigned char smemRaw[4 * MROWS * A_STRIDE * 2];
  __shared__ float Bsh[NEXP];

  unsigned short* AH = (unsigned short*)smemRaw;       // [2][MROWS*A_STRIDE]
  unsigned short* AL = AH + 2 * MROWS * A_STRIDE;      // [2][MROWS*A_STRIDE]
  float* Ssc = (float*)smemRaw;                        // [MROWS*SC_STRIDE]

  const int tid  = threadIdx.x;
  const int wave = tid >> 5;
  const int lane = tid & 31;
  const int m0   = blockIdx.x * MROWS;

  Bsh[tid] = bias[tid];   // blockDim == 256 == NEXP

  // ---------------- GEMM: logits[32 x 256] ----------------
  const int e0     = wave * 32;          // this wave's expert group
  const int nlane  = lane & 15;
  const int khalfA = (lane >> 4) << 3;   // A frag: K {0..7,16..23}/{8..15,24..31}
  const int khalfB = (lane >> 4) << 3;   // B frag: K {0..7,16..23}/{8..15,24..31}

  v8f acc00 = {0.f, 0.f, 0.f, 0.f, 0.f, 0.f, 0.f, 0.f};  // (mtile0, ntile0)
  v8f acc01 = acc00, acc10 = acc00, acc11 = acc00;

  const unsigned bOff0 = (unsigned)(e0 + nlane) * HDIM + khalfB;
  const unsigned bOff1 = bOff0 + 16u * HDIM;

  // staging: thread -> (row, 16 consecutive k)
  const int stRow = tid >> 3;
  const int stCol = (tid & 7) << 4;
  const float* xPtr = x + (size_t)(m0 + stRow) * HDIM + stCol;

  float4 p0 = *reinterpret_cast<const float4*>(xPtr);
  float4 p1 = *reinterpret_cast<const float4*>(xPtr + 4);
  float4 p2 = *reinterpret_cast<const float4*>(xPtr + 8);
  float4 p3 = *reinterpret_cast<const float4*>(xPtr + 12);

  int buf = 0;
  for (int kb = 0; kb < HDIM; kb += KC) {
    {  // convert current chunk to bf16 hi/lo, store to LDS[buf]
      float f[16] = {p0.x, p0.y, p0.z, p0.w, p1.x, p1.y, p1.z, p1.w,
                     p2.x, p2.y, p2.z, p2.w, p3.x, p3.y, p3.z, p3.w};
      unsigned short hv[16], lv[16];
#pragma unroll
      for (int j = 0; j < 16; ++j) {
        hv[j] = f32_to_bf16_rne(f[j]);
        lv[j] = f32_to_bf16_rne(f[j] - bf16_bits_to_f32(hv[j]));
      }
      const int sBase = buf * (MROWS * A_STRIDE) + stRow * A_STRIDE + stCol;
      uint4 Ha = {(unsigned)hv[0] | ((unsigned)hv[1] << 16),
                  (unsigned)hv[2] | ((unsigned)hv[3] << 16),
                  (unsigned)hv[4] | ((unsigned)hv[5] << 16),
                  (unsigned)hv[6] | ((unsigned)hv[7] << 16)};
      uint4 Hb = {(unsigned)hv[8] | ((unsigned)hv[9] << 16),
                  (unsigned)hv[10] | ((unsigned)hv[11] << 16),
                  (unsigned)hv[12] | ((unsigned)hv[13] << 16),
                  (unsigned)hv[14] | ((unsigned)hv[15] << 16)};
      uint4 La = {(unsigned)lv[0] | ((unsigned)lv[1] << 16),
                  (unsigned)lv[2] | ((unsigned)lv[3] << 16),
                  (unsigned)lv[4] | ((unsigned)lv[5] << 16),
                  (unsigned)lv[6] | ((unsigned)lv[7] << 16)};
      uint4 Lb = {(unsigned)lv[8] | ((unsigned)lv[9] << 16),
                  (unsigned)lv[10] | ((unsigned)lv[11] << 16),
                  (unsigned)lv[12] | ((unsigned)lv[13] << 16),
                  (unsigned)lv[14] | ((unsigned)lv[15] << 16)};
      *reinterpret_cast<uint4*>(AH + sBase)     = Ha;
      *reinterpret_cast<uint4*>(AH + sBase + 8) = Hb;
      *reinterpret_cast<uint4*>(AL + sBase)     = La;
      *reinterpret_cast<uint4*>(AL + sBase + 8) = Lb;
    }
    __syncthreads();

    // prefetch next chunk into registers (latency hidden by 48 WMMAs)
    if (kb + KC < HDIM) {
      p0 = *reinterpret_cast<const float4*>(xPtr + kb + KC);
      p1 = *reinterpret_cast<const float4*>(xPtr + kb + KC + 4);
      p2 = *reinterpret_cast<const float4*>(xPtr + kb + KC + 8);
      p3 = *reinterpret_cast<const float4*>(xPtr + kb + KC + 12);
    }

    const unsigned short* aB =
        AH + buf * (MROWS * A_STRIDE) + nlane * A_STRIDE + khalfA;
    const unsigned short* lB =
        AL + buf * (MROWS * A_STRIDE) + nlane * A_STRIDE + khalfA;
#pragma unroll
    for (int ks = 0; ks < KC; ks += 32) {
      v16bf ah0 = cat8(lds8(aB + ks), lds8(aB + ks + 16));
      v16bf ah1 = cat8(lds8(aB + 16 * A_STRIDE + ks),
                       lds8(aB + 16 * A_STRIDE + ks + 16));
      v16bf al0 = cat8(lds8(lB + ks), lds8(lB + ks + 16));
      v16bf al1 = cat8(lds8(lB + 16 * A_STRIDE + ks),
                       lds8(lB + 16 * A_STRIDE + ks + 16));
      const unsigned ko = (unsigned)(kb + ks);
      v16bf bh0 = cat8(ldg8(wh, bOff0 + ko), ldg8(wh, bOff0 + ko + 16));
      v16bf bh1 = cat8(ldg8(wh, bOff1 + ko), ldg8(wh, bOff1 + ko + 16));
      v16bf bl0 = cat8(ldg8(wl, bOff0 + ko), ldg8(wl, bOff0 + ko + 16));
      v16bf bl1 = cat8(ldg8(wl, bOff1 + ko), ldg8(wl, bOff1 + ko + 16));
      // 12 WMMAs, accumulator dependency distance 4
      acc00 = wmma_bf16(ah0, bh0, acc00);   // xh*wh
      acc01 = wmma_bf16(ah0, bh1, acc01);
      acc10 = wmma_bf16(ah1, bh0, acc10);
      acc11 = wmma_bf16(ah1, bh1, acc11);
      acc00 = wmma_bf16(al0, bh0, acc00);   // xl*wh
      acc01 = wmma_bf16(al0, bh1, acc01);
      acc10 = wmma_bf16(al1, bh0, acc10);
      acc11 = wmma_bf16(al1, bh1, acc11);
      acc00 = wmma_bf16(ah0, bl0, acc00);   // xh*wl
      acc01 = wmma_bf16(ah0, bl1, acc01);
      acc10 = wmma_bf16(ah1, bl0, acc10);
      acc11 = wmma_bf16(ah1, bl1, acc11);
    }
    buf ^= 1;
  }
  __syncthreads();   // A staging dead; score tile aliases that LDS now

  // ---------------- sigmoid + bias -> LDS scores ----------------
  const int mHalf = (lane >> 4) << 3;   // C layout: lanes 16..31 hold M+8
#pragma unroll
  for (int r = 0; r < 8; ++r) {
    int ec0 = e0 + nlane;
    int ec1 = e0 + 16 + nlane;
    int mmA = r + mHalf;          // m-tile 0
    int mmB = 16 + r + mHalf;     // m-tile 1
    float sA0 = 1.0f / (1.0f + __expf(-acc00[r]));
    float sA1 = 1.0f / (1.0f + __expf(-acc01[r]));
    float sB0 = 1.0f / (1.0f + __expf(-acc10[r]));
    float sB1 = 1.0f / (1.0f + __expf(-acc11[r]));
    Ssc[mmA * SC_STRIDE + ec0] = sA0 + Bsh[ec0];
    Ssc[mmA * SC_STRIDE + ec1] = sA1 + Bsh[ec1];
    Ssc[mmB * SC_STRIDE + ec0] = sB0 + Bsh[ec0];
    Ssc[mmB * SC_STRIDE + ec1] = sB1 + Bsh[ec1];
  }
  __syncthreads();

  // ---------------- grouped top-k (4 rows per wave) ----------------
  const float NEG = -__builtin_inff();
  for (int rr = 0; rr < 4; ++rr) {
    const int m = wave * 4 + rr;
    const float* rowp = &Ssc[m * SC_STRIDE];

    // group top-2: 4 lanes per group, 8 experts per lane, butterfly merge
    float m1 = NEG, m2 = NEG;
    {
      int base = (lane >> 2) * 32 + (lane & 3) * 8;
#pragma unroll
      for (int i = 0; i < 8; ++i) {
        float s = rowp[base + i];
        if (s > m1)      { m2 = m1; m1 = s; }
        else if (s > m2) { m2 = s; }
      }
    }
#pragma unroll
    for (int off = 1; off <= 2; off <<= 1) {
      float o1 = __shfl_xor(m1, off, 32);
      float o2 = __shfl_xor(m2, off, 32);
      if (o1 > m1) { m2 = fmaxf(m1, o2); m1 = o1; }
      else         { m2 = fmaxf(m2, o1); }
    }
    float gsum = m1 + m2;

    float gs[8];
#pragma unroll
    for (int j = 0; j < 8; ++j) gs[j] = __shfl(gsum, j * 4, 32);

    unsigned gmask = 0;
#pragma unroll
    for (int t = 0; t < 4; ++t) {
      float best = NEG; int bi = 0;
#pragma unroll
      for (int j = 0; j < 8; ++j)
        if (!((gmask >> j) & 1u) && gs[j] > best) { best = gs[j]; bi = j; }
      gmask |= (1u << bi);
    }

    float v[8];
#pragma unroll
    for (int j = 0; j < 8; ++j) {
      float s = rowp[j * 32 + lane];
      v[j] = ((gmask >> j) & 1u) ? s : NEG;
    }
    int widx[8]; float wval[8];
#pragma unroll
    for (int i = 0; i < 8; ++i) {
      float lv = NEG; int lj = 0;
#pragma unroll
      for (int j = 0; j < 8; ++j)
        if (v[j] > lv) { lv = v[j]; lj = j; }
      float bv = lv; int be = lj * 32 + lane;
#pragma unroll
      for (int off = 16; off >= 1; off >>= 1) {
        float ov = __shfl_xor(bv, off, 32);
        int   oe = __shfl_xor(be, off, 32);
        if (ov > bv || (ov == bv && oe < be)) { bv = ov; be = oe; }
      }
      widx[i] = be; wval[i] = bv;
#pragma unroll
      for (int j = 0; j < 8; ++j)
        if (be == j * 32 + lane) v[j] = NEG;
    }

    float denom = 1e-20f;
    float uw[8];
#pragma unroll
    for (int i = 0; i < 8; ++i) {
      uw[i] = wval[i] - Bsh[widx[i]];
      denom += uw[i];
    }
    float scl = 2.5f / denom;
    if (lane == 0) {
      int base = (m0 + m) * 8;
#pragma unroll
      for (int i = 0; i < 8; ++i) {
        outIdx[base + i] = (float)widx[i];
        outW[base + i]   = uw[i] * scl;
      }
    }
  }
}

extern "C" void kernel_launch(void* const* d_in, const int* in_sizes, int n_in,
                              void* d_out, int out_size, void* d_ws, size_t ws_size,
                              hipStream_t stream) {
  const float* x    = (const float*)d_in[0];   // [T, 7168]
  const float* w    = (const float*)d_in[1];   // [256, 7168]
  const float* bias = (const float*)d_in[2];   // [256]
  const int T = in_sizes[0] / HDIM;            // 8192

  unsigned short* wh = (unsigned short*)d_ws;              // [256*7168]
  unsigned short* wl = wh + (size_t)NEXP * HDIM;           // [256*7168]

  float* outIdx = (float*)d_out;               // [T*8] indices (as float)
  float* outW   = outIdx + (size_t)T * 8;      // [T*8] weights

  const int splitBlocks = (NEXP * HDIM) / (256 * 4);       // 1792
  hipLaunchKernelGGL(split_w_kernel, dim3(splitBlocks), dim3(256), 0, stream,
                     w, wh, wl);

  const int blocks = T / MROWS;                // 256
  hipLaunchKernelGGL(moe_gate_kernel, dim3(blocks), dim3(256), 0, stream,
                     x, wh, wl, bias, outIdx, outW);
}
